// baseGNN_79293686218936
// MI455X (gfx1250) — matrix operand order repaired and checked by reading, not compiled, not run on hardware
//
#include <hip/hip_runtime.h>

// ---------------------------------------------------------------------------
// GNN forward for MI455X (gfx1250, wave32).
//  - GEMMs: v_wmma_f32_16x16x32_bf16, operands staged in LDS in exact
//    per-lane fragment layout (ds_load_b128 feeds wmma directly).
//  - Weights pre-swizzled to bf16 fragments once per layer; each block pulls
//    them with global_load_async_to_lds_b128 (ASYNCcnt), contiguous 32 KB.
//  - BatchNorm of the previous layer is fused into the GEMM A-staging
//    (applied during f32->bf16 conversion); self-loop term fused into the
//    GEMM epilogue; final-layer BN fused into the pooling scatter.
//  - Edge aggregation: native f32 global atomics, one wave per edge.
// ---------------------------------------------------------------------------

#define N_NODES 50000
#define N_EDGES 800000
#define DIMH    128
#define NGRAPH  100
#define NCLASS  2
#define BN_EPS  1e-5f

typedef __attribute__((ext_vector_type(16))) __bf16 v16bf;
typedef __attribute__((ext_vector_type(8)))  float  v8f;

// ---------------- W pre-swizzle: f32 [128,128] -> bf16 fragment stream ------
// Element f = ((ct*4 + kstep)*32 + lane)*16 + j  holds
//   bf16( W[(kstep*32 + (lane>=16 ? 16:0) + j) * 128 + ct*16 + (lane&15)] )
// i.e. B fragments for v_wmma_f32_16x16x32_bf16 in contiguous k order (k = 16*half + j), as A is staged.
__global__ void swizzle_w_kernel(const float* __restrict__ W,
                                 unsigned short* __restrict__ Wsw) {
  int f = blockIdx.x * blockDim.x + threadIdx.x;     // 16384 total
  int j    = f & 15;
  int lane = (f >> 4) & 31;
  int kf   = f >> 9;                                 // ct*4 + kstep
  int kstep = kf & 3, ct = kf >> 2;
  int K   = kstep * 32 + ((lane >> 4) * 16) + j;
  int col = ct * 16 + (lane & 15);
  __bf16 b = (__bf16)W[K * DIMH + col];
  Wsw[f] = __builtin_bit_cast(unsigned short, b);
}

// per-channel batchnorm on a float4 whose first channel is c
__device__ __forceinline__ float4 bn4(float4 v, const float* __restrict__ g,
                                      const float* __restrict__ be,
                                      const float* __restrict__ mu,
                                      const float* __restrict__ rs, int c) {
  float4 r;
  r.x = g[c + 0] * (v.x - mu[c + 0]) * rs[c + 0] + be[c + 0];
  r.y = g[c + 1] * (v.y - mu[c + 1]) * rs[c + 1] + be[c + 1];
  r.z = g[c + 2] * (v.z - mu[c + 2]) * rs[c + 2] + be[c + 2];
  r.w = g[c + 3] * (v.w - mu[c + 3]) * rs[c + 3] + be[c + 3];
  return r;
}

// ---------------- dense GEMM: H = bn(Z) @ W, Agg = H * dinv^2 ---------------
// 128 threads = 4 waves per block; block covers rows [16b,16b+16).
// Wave w owns column tiles 2w and 2w+1 (two independent accumulators).
// BN==true: apply previous layer's batchnorm while converting A to bf16.
template <bool BN>
__global__ __launch_bounds__(128) void gemm_wmma_kernel(
    const float* __restrict__ Z, const unsigned short* __restrict__ Wsw,
    const float* __restrict__ gvec, const float* __restrict__ bevec,
    const float* __restrict__ stats, const float* __restrict__ dinv,
    float* __restrict__ Hout, float* __restrict__ Agg) {
  __shared__ __align__(32) __bf16 sB[8 * 4 * 32 * 16];   // 32 KB: B fragments
  __shared__ __align__(32) __bf16 sA[4 * 32 * 16];       //  4 KB: A fragments

  const int tid = threadIdx.x;
  const int r0  = blockIdx.x * 16;

  // ---- async copy of pre-swizzled W fragments into LDS (contiguous) ----
  {
    unsigned lds = (unsigned)(uintptr_t)&sB[tid * 8];    // 16 B per thread
    const unsigned short* gp = Wsw + tid * 8;
    #pragma unroll
    for (int i = 0; i < 16; ++i) {                       // 128thr*16*16B = 32KB
      asm volatile("global_load_async_to_lds_b128 %0, %1, off"
                   :: "v"(lds), "v"(gp) : "memory");
      lds += 2048;
      gp  += 1024;
    }
  }

  // ---- stage A tile (16 rows) as bf16 fragments: thread t -> fragment t ----
  {
    int lane2 = tid & 31, ks = tid >> 5;                 // 4 ksteps * 32 lanes
    int base = ks * 32 + ((lane2 >> 4) * 16);            // first K channel
    const float* zr = Z + (size_t)(r0 + (lane2 & 15)) * DIMH + base;
    const float4* zp = (const float4*)zr;
    float4 f0 = zp[0], f1 = zp[1];                       // channels base+0..7
    float4 f2 = zp[2], f3 = zp[3];                       // channels base+8..15
    if (BN) {
      const float* mu = stats + 2 * DIMH;
      const float* rs = stats + 3 * DIMH;
      f0 = bn4(f0, gvec, bevec, mu, rs, base + 0);
      f1 = bn4(f1, gvec, bevec, mu, rs, base + 4);
      f2 = bn4(f2, gvec, bevec, mu, rs, base + 8);
      f3 = bn4(f3, gvec, bevec, mu, rs, base + 12);
    }
    v16bf Av;
    Av[0]=(__bf16)f0.x;  Av[1]=(__bf16)f0.y;  Av[2]=(__bf16)f0.z;  Av[3]=(__bf16)f0.w;
    Av[4]=(__bf16)f1.x;  Av[5]=(__bf16)f1.y;  Av[6]=(__bf16)f1.z;  Av[7]=(__bf16)f1.w;
    Av[8]=(__bf16)f2.x;  Av[9]=(__bf16)f2.y;  Av[10]=(__bf16)f2.z; Av[11]=(__bf16)f2.w;
    Av[12]=(__bf16)f3.x; Av[13]=(__bf16)f3.y; Av[14]=(__bf16)f3.z; Av[15]=(__bf16)f3.w;
    *(v16bf*)&sA[tid * 16] = Av;
  }

  asm volatile("s_wait_asynccnt 0" ::: "memory");        // W fragments landed
  __syncthreads();

  const int wave = tid >> 5;
  const int lane = tid & 31;
  const int half = lane >> 4;
  const int n    = lane & 15;
  const int ct0  = wave * 2, ct1 = wave * 2 + 1;

  v8f acc0 = {}, acc1 = {};
  #pragma unroll
  for (int ks = 0; ks < 4; ++ks) {
    v16bf A  = *(const v16bf*)&sA[(ks * 32 + lane) * 16];
    v16bf B0 = *(const v16bf*)&sB[((ct0 * 4 + ks) * 32 + lane) * 16];
    v16bf B1 = *(const v16bf*)&sB[((ct1 * 4 + ks) * 32 + lane) * 16];
    acc0 = __builtin_amdgcn_wmma_f32_16x16x32_bf16(
        false, A, false, B0, (short)0, acc0, false, false);
    acc1 = __builtin_amdgcn_wmma_f32_16x16x32_bf16(
        false, A, false, B1, (short)0, acc1, false, false);
  }

  // D layout: VGPR r -> M = r + 8*half, N = lane&15.
  // Fused epilogue: h -> Hout, self-loop term h*dinv^2 -> Agg.
  #pragma unroll
  for (int r = 0; r < 8; ++r) {
    const int grow = r0 + r + half * 8;
    const float di = dinv[grow];
    const float d2 = di * di;
    const size_t row = (size_t)grow * DIMH;
    const float h0 = acc0[r], h1 = acc1[r];
    Hout[row + ct0 * 16 + n] = h0;
    Hout[row + ct1 * 16 + n] = h1;
    Agg[row + ct0 * 16 + n] = h0 * d2;
    Agg[row + ct1 * 16 + n] = h1 * d2;
  }
}

// ---------------- utility fill ---------------------------------------------
__global__ void fill_kernel(float* p, float v, int n) {
  int i = blockIdx.x * blockDim.x + threadIdx.x;
  if (i < n) p[i] = v;
}

// ---------------- degree accumulation & dinv -------------------------------
__global__ void deg_edges_kernel(const int* __restrict__ ei,
                                 const float* __restrict__ ea,
                                 float* __restrict__ deg) {
  int e = blockIdx.x * blockDim.x + threadIdx.x;
  if (e >= N_EDGES) return;
  unsafeAtomicAdd(&deg[ei[N_EDGES + e]], fabsf(ea[e]));
}

__global__ void deg_to_dinv_kernel(float* __restrict__ deg) {
  int i = blockIdx.x * blockDim.x + threadIdx.x;
  if (i >= N_NODES) return;
  float d = deg[i];
  deg[i] = (d > 0.0f) ? rsqrtf(d) : 0.0f;
}

// ---------------- edge scatter: one wave per edge --------------------------
__global__ __launch_bounds__(256) void edge_scatter_kernel(
    const int* __restrict__ ei, const float* __restrict__ ea,
    const float* __restrict__ dinv, const float* __restrict__ H,
    float* __restrict__ Agg) {
  int gw = (blockIdx.x * 256 + threadIdx.x) >> 5;    // global wave id
  int lane = threadIdx.x & 31;
  if (gw >= N_EDGES) return;
  int s = ei[gw];
  int d = ei[N_EDGES + gw];
  float norm = dinv[s] * fabsf(ea[gw]) * dinv[d];
  const float4* hp = (const float4*)(H + (size_t)s * DIMH);
  float4 v = hp[lane];                               // 32 lanes x 4 ch = 128
  float* out = Agg + (size_t)d * DIMH + lane * 4;
  unsafeAtomicAdd(out + 0, v.x * norm);
  unsafeAtomicAdd(out + 1, v.y * norm);
  unsafeAtomicAdd(out + 2, v.z * norm);
  unsafeAtomicAdd(out + 3, v.w * norm);
}

// ---------------- bias + leaky-relu + batch statistics ---------------------
__global__ __launch_bounds__(128) void bias_leaky_stats_kernel(
    float* __restrict__ Z, const float* __restrict__ b,
    float* __restrict__ stats /* [0..127]=sum, [128..255]=sumsq */) {
  int c = threadIdx.x;
  float bc = b[c];
  float s1 = 0.0f, s2 = 0.0f;
  for (int n = blockIdx.x; n < N_NODES; n += gridDim.x) {
    if (n + (int)gridDim.x < N_NODES)
      __builtin_prefetch(&Z[(size_t)(n + gridDim.x) * DIMH + c], 0, 0);
    float v = Z[(size_t)n * DIMH + c] + bc;
    v = (v > 0.0f) ? v : 0.2f * v;                   // leaky_relu(0.2)
    Z[(size_t)n * DIMH + c] = v;
    s1 += v;
    s2 += v * v;
  }
  unsafeAtomicAdd(&stats[c], s1);
  unsafeAtomicAdd(&stats[DIMH + c], s2);
}

__global__ void bn_finalize_kernel(float* __restrict__ stats) {
  int c = threadIdx.x;                               // 128 threads
  float mu  = stats[c] * (1.0f / N_NODES);
  float var = stats[DIMH + c] * (1.0f / N_NODES) - mu * mu;
  stats[2 * DIMH + c] = mu;
  stats[3 * DIMH + c] = rsqrtf(var + BN_EPS);
}

// ---------------- mean pooling per graph (final-layer BN fused) ------------
__global__ void pool_scatter_kernel(const float* __restrict__ Z,
                                    const int* __restrict__ batch,
                                    const float* __restrict__ g,
                                    const float* __restrict__ be,
                                    const float* __restrict__ stats,
                                    float* __restrict__ pooled,
                                    float* __restrict__ counts) {
  int idx = blockIdx.x * blockDim.x + threadIdx.x;   // N*128 exact
  int node = idx >> 7, c = idx & (DIMH - 1);
  float mu = stats[2 * DIMH + c];
  float rs = stats[3 * DIMH + c];
  float v = g[c] * (Z[idx] - mu) * rs + be[c];
  int gi = batch[node];
  unsafeAtomicAdd(&pooled[gi * DIMH + c], v);
  if (c == 0) unsafeAtomicAdd(&counts[gi], 1.0f);
}

__global__ void pool_div_kernel(float* __restrict__ pooled,
                                const float* __restrict__ counts) {
  int idx = blockIdx.x * blockDim.x + threadIdx.x;
  if (idx >= NGRAPH * DIMH) return;
  pooled[idx] /= fmaxf(counts[idx >> 7], 1.0f);
}

// ---------------- classifier MLP (tiny: 100 rows) --------------------------
__global__ void mlp1_kernel(const float* __restrict__ pooled,
                            const float* __restrict__ Wm1,
                            const float* __restrict__ bm1,
                            float* __restrict__ hidden) {
  int idx = blockIdx.x * blockDim.x + threadIdx.x;
  if (idx >= NGRAPH * DIMH) return;
  int gi = idx >> 7, j = idx & (DIMH - 1);
  float acc = bm1[j];
  for (int k = 0; k < DIMH; ++k) acc += pooled[gi * DIMH + k] * Wm1[k * DIMH + j];
  hidden[idx] = fmaxf(acc, 0.0f);
}

__global__ void mlp2_kernel(const float* __restrict__ hidden,
                            const float* __restrict__ Wm2,
                            const float* __restrict__ bm2,
                            float* __restrict__ out) {
  int idx = blockIdx.x * blockDim.x + threadIdx.x;
  if (idx >= NGRAPH * NCLASS) return;
  int gi = idx / NCLASS, c = idx % NCLASS;
  float acc = bm2[c];
  for (int k = 0; k < DIMH; ++k) acc += hidden[gi * DIMH + k] * Wm2[k * NCLASS + c];
  out[idx] = acc;
}

// ---------------------------------------------------------------------------
extern "C" void kernel_launch(void* const* d_in, const int* in_sizes, int n_in,
                              void* d_out, int out_size, void* d_ws, size_t ws_size,
                              hipStream_t stream) {
  const float* x     = (const float*)d_in[0];
  const int*   ei    = (const int*)d_in[1];
  const float* ea    = (const float*)d_in[2];
  const int*   batch = (const int*)d_in[3];
  const float* W[3]  = {(const float*)d_in[4],  (const float*)d_in[8],  (const float*)d_in[12]};
  const float* b[3]  = {(const float*)d_in[5],  (const float*)d_in[9],  (const float*)d_in[13]};
  const float* g[3]  = {(const float*)d_in[6],  (const float*)d_in[10], (const float*)d_in[14]};
  const float* be[3] = {(const float*)d_in[7],  (const float*)d_in[11], (const float*)d_in[15]};
  const float* Wm1 = (const float*)d_in[16];
  const float* bm1 = (const float*)d_in[17];
  const float* Wm2 = (const float*)d_in[18];
  const float* bm2 = (const float*)d_in[19];
  float* out = (float*)d_out;

  // workspace carve-up
  float* ws     = (float*)d_ws;
  const size_t NH = (size_t)N_NODES * DIMH;
  float* bufA   = ws;                       // z / agg buffer  (N*128)
  float* bufB   = bufA + NH;                // h buffer        (N*128)
  float* dinv   = bufB + NH;                // deg -> dinv     (N)
  float* stats  = dinv + N_NODES;           // sum|sumsq|mu|rstd (4*128)
  float* pooled = stats + 4 * DIMH;         // (G*128)
  float* counts = pooled + NGRAPH * DIMH;   // (G), contiguous with pooled
  float* hidden = counts + NGRAPH;          // (G*128)
  unsigned short* Wsw = (unsigned short*)(hidden + NGRAPH * DIMH); // 16384 u16

  const int elemBlocks = (N_NODES * DIMH) / 256;   // 25000, exact
  const int edgeWaveBlocks = N_EDGES / 8;          // 100000: 8 waves/block

  // --- symmetric normalization terms (layer-independent) ---
  fill_kernel<<<(N_NODES + 255) / 256, 256, 0, stream>>>(dinv, 1.0f, N_NODES);
  deg_edges_kernel<<<(N_EDGES + 255) / 256, 256, 0, stream>>>(ei, ea, dinv);
  deg_to_dinv_kernel<<<(N_NODES + 255) / 256, 256, 0, stream>>>(dinv);

  // --- 3 GCN blocks ---
  for (int l = 0; l < 3; ++l) {
    swizzle_w_kernel<<<(DIMH * DIMH) / 256, 256, 0, stream>>>(W[l], Wsw);
    if (l == 0) {
      // input is raw x, no batchnorm on the A operand
      gemm_wmma_kernel<false><<<N_NODES / 16, 128, 0, stream>>>(
          x, Wsw, nullptr, nullptr, nullptr, dinv, bufB, bufA);
    } else {
      // apply previous layer's batchnorm while staging A (reads stats of l-1)
      gemm_wmma_kernel<true><<<N_NODES / 16, 128, 0, stream>>>(
          bufA, Wsw, g[l - 1], be[l - 1], stats, dinv, bufB, bufA);
    }
    edge_scatter_kernel<<<edgeWaveBlocks, 256, 0, stream>>>(ei, ea, dinv, bufB, bufA);
    fill_kernel<<<1, 256, 0, stream>>>(stats, 0.0f, 2 * DIMH);
    bias_leaky_stats_kernel<<<512, 128, 0, stream>>>(bufA, b[l], stats);
    bn_finalize_kernel<<<1, 128, 0, stream>>>(stats);
  }

  // --- mean pool (fuses layer-2 batchnorm) + MLP head ---
  fill_kernel<<<(NGRAPH * DIMH + NGRAPH + 255) / 256, 256, 0, stream>>>(
      pooled, 0.0f, NGRAPH * DIMH + NGRAPH);
  pool_scatter_kernel<<<elemBlocks, 256, 0, stream>>>(
      bufA, batch, g[2], be[2], stats, pooled, counts);
  pool_div_kernel<<<(NGRAPH * DIMH + 255) / 256, 256, 0, stream>>>(pooled, counts);
  mlp1_kernel<<<(NGRAPH * DIMH + 255) / 256, 256, 0, stream>>>(pooled, Wm1, bm1, hidden);
  mlp2_kernel<<<1, 256, 0, stream>>>(hidden, Wm2, bm2, out);
}
